// Sggnn_gcn_32693291057656
// MI455X (gfx1250) — matrix operand order repaired and checked by reading, not compiled, not run on hardware
//
#include <hip/hip_runtime.h>
#include <hip/hip_bf16.h>

typedef __attribute__((ext_vector_type(16))) __bf16 v16bf;
typedef __attribute__((ext_vector_type(8)))  float  v8f;

#define BN_EPS 1e-5f
#define SLOPE  0.1f
#define LDH    520   // padded LDS row stride in halves

union FragBF {
  v16bf v;
  unsigned short u[16];
  uint4 q[2];
};

// native f32 -> bf16 (RNE); bf16 is a legal first-class type on gfx1250,
// so let the backend pick v_cvt_pk_bf16_f32 instead of a 3-op emulation.
__device__ __forceinline__ unsigned short f2bf(float f) {
  __bf16 b = (__bf16)f;
  return __builtin_bit_cast(unsigned short, b);
}

__device__ __forceinline__ unsigned int pk2bf(float lo, float hi) {
  return (unsigned int)f2bf(lo) | ((unsigned int)f2bf(hi) << 16);
}

__device__ __forceinline__ v8f vzero() {
  v8f z;
#pragma unroll
  for (int i = 0; i < 8; ++i) z[i] = 0.0f;
  return z;
}

// A-fragment (16x32 bf16): per lane, 16 contiguous K at base = (lane/16)*16
__device__ __forceinline__ v16bf fragA_bf16(const unsigned short* p) {
  FragBF fr;
  fr.q[0] = *(const uint4*)(p);
  fr.q[1] = *(const uint4*)(p + 8);
  return fr.v;
}
// B-fragment (32x16 bf16): per lane, 16 contiguous K at base = (lane/16)*16
__device__ __forceinline__ v16bf fragB_bf16(const unsigned short* p) {
  FragBF fr;
  fr.q[0] = *(const uint4*)(p);
  fr.q[1] = *(const uint4*)(p + 8);
  return fr.v;
}

// ---------------- adjacency normalization ----------------
__global__ void k_dinv(const float* __restrict__ w, float* __restrict__ dinv) {
  int row = blockIdx.x;           // 512 rows, one wave32 per row
  int lane = threadIdx.x;
  const float* wr = w + row * 512;
  float s = 0.f;
  for (int j = lane; j < 512; j += 32) s += wr[j];
#pragma unroll
  for (int m = 16; m > 0; m >>= 1) s += __shfl_xor(s, m, 32);
  s += 1.0f;                      // + identity on the diagonal
  float r = (s > 0.f) ? rsqrtf(s) : 0.f;
  if (lane == 0) dinv[row] = r;
}

// wnf[m][n] = w_norm[m][n] = adj[n][m] * dinv[m] * dinv[n]  (f32, natural orientation)
__global__ void k_wnorm(const float* __restrict__ w, const float* __restrict__ dinv,
                        float* __restrict__ wnf) {
  int idx = blockIdx.x * blockDim.x + threadIdx.x;   // 512*512
  int m = idx >> 9, n = idx & 511;
  float adj = w[n * 512 + m] + (n == m ? 1.0f : 0.0f);
  wnf[idx] = adj * dinv[m] * dinv[n];
}

// f32 -> bf16 bulk convert (vectorized by 4)
__global__ void k_cvt(const float* __restrict__ s, unsigned short* __restrict__ d, int n4) {
  int i = blockIdx.x * blockDim.x + threadIdx.x;
  if (i < n4) {
    float4 f = ((const float4*)s)[i];
    ((uint2*)d)[i] = make_uint2(pk2bf(f.x, f.y), pk2bf(f.z, f.w));
  }
}

// ---------------- fused MLP x2 + classifier projection: u = lrelu(bn2(...)) @ Wc^T ----
__global__ void __launch_bounds__(256) k_mlp(
    const float* __restrict__ x,                       // 65536 x 512 (f32)
    const unsigned short* __restrict__ W1b,            // 512 x 512 bf16
    const float* __restrict__ b1,
    const float* __restrict__ g1, const float* __restrict__ be1,
    const float* __restrict__ m1, const float* __restrict__ v1,
    const unsigned short* __restrict__ W2b,            // 512 x 512 bf16
    const float* __restrict__ b2,
    const float* __restrict__ g2, const float* __restrict__ be2,
    const float* __restrict__ m2, const float* __restrict__ v2,
    const float* __restrict__ Wc,                      // [2][512]
    float* __restrict__ u)                             // [65536][2] f32
{
  extern __shared__ unsigned short smem[];
  unsigned short* xl = smem;               // [64][LDH] bf16 x-tile (stage 0/1)
  unsigned short* hl = smem + 64 * LDH;    // [64][LDH] bf16 h-tile
  float* up = (float*)smem;                // [8][64][2] partials (reuses xl after stage 1)

  const int tid  = threadIdx.x;
  const int wv   = tid >> 5;               // 8 waves: 64 cols each
  const int lane = tid & 31;
  const int hf   = lane >> 4;
  const int cl   = lane & 15;
  const int r0   = blockIdx.x * 64;        // 64 rows per block

  // ---- stage 0: cooperative load+convert of the x tile (once per block)
  {
    int m  = tid >> 2;                     // 4 threads per row
    int k0 = (tid & 3) * 128;
    const float4* src = (const float4*)(x + (size_t)(r0 + m) * 512 + k0);
    uint2* dst = (uint2*)(xl + m * LDH + k0);
#pragma unroll
    for (int i = 0; i < 32; ++i) {
      float4 f = src[i];
      dst[i] = make_uint2(pk2bf(f.x, f.y), pk2bf(f.z, f.w));
    }
  }
  __syncthreads();

  // ---- stage 1: h = lrelu(bn1(x @ W1^T + b1)) -> LDS (bf16)
  {
    v8f acc[4][4];
#pragma unroll
    for (int i = 0; i < 4; ++i)
#pragma unroll
      for (int j = 0; j < 4; ++j) acc[i][j] = vzero();

    for (int kb = 0; kb < 512; kb += 32) {
      v16bf a[4], bfr[4];
#pragma unroll
      for (int mt = 0; mt < 4; ++mt)
        a[mt] = fragA_bf16(xl + (mt * 16 + cl) * LDH + kb + hf * 16);
#pragma unroll
      for (int nt = 0; nt < 4; ++nt) {
        int n = wv * 64 + nt * 16 + cl;
        bfr[nt] = fragB_bf16(W1b + (size_t)n * 512 + kb + hf * 16);
      }
#pragma unroll
      for (int mt = 0; mt < 4; ++mt)
#pragma unroll
        for (int nt = 0; nt < 4; ++nt)
          acc[mt][nt] = __builtin_amdgcn_wmma_f32_16x16x32_bf16(
              false, a[mt], false, bfr[nt], (short)0, acc[mt][nt], false, false);
    }

#pragma unroll
    for (int nt = 0; nt < 4; ++nt) {
      int n = wv * 64 + nt * 16 + cl;
      float s   = g1[n] * rsqrtf(v1[n] + BN_EPS);
      float off = b1[n] - m1[n];
      float bt  = be1[n];
#pragma unroll
      for (int mt = 0; mt < 4; ++mt) {
#pragma unroll
        for (int vv = 0; vv < 8; ++vv) {
          float y = (acc[mt][nt][vv] + off) * s + bt;
          y = y > 0.f ? y : SLOPE * y;
          hl[(mt * 16 + vv + 8 * hf) * LDH + n] = f2bf(y);
        }
      }
    }
  }
  __syncthreads();

  // ---- stage 2: t = lrelu(bn2(h @ W2^T + b2)); fused u = t @ Wc^T (f32, in regs)
  {
    v8f acc[4][4];
#pragma unroll
    for (int i = 0; i < 4; ++i)
#pragma unroll
      for (int j = 0; j < 4; ++j) acc[i][j] = vzero();

    for (int kb = 0; kb < 512; kb += 32) {
      v16bf a[4], bfr[4];
#pragma unroll
      for (int mt = 0; mt < 4; ++mt)
        a[mt] = fragA_bf16(hl + (mt * 16 + cl) * LDH + kb + hf * 16);
#pragma unroll
      for (int nt = 0; nt < 4; ++nt) {
        int f = wv * 64 + nt * 16 + cl;
        bfr[nt] = fragB_bf16(W2b + (size_t)f * 512 + kb + hf * 16);
      }
#pragma unroll
      for (int mt = 0; mt < 4; ++mt)
#pragma unroll
        for (int nt = 0; nt < 4; ++nt)
          acc[mt][nt] = __builtin_amdgcn_wmma_f32_16x16x32_bf16(
              false, a[mt], false, bfr[nt], (short)0, acc[mt][nt], false, false);
    }

    // per-nt BN constants and classifier weights for this lane's f
    float s[4], off[4], bt[4], wc0[4], wc1[4];
#pragma unroll
    for (int nt = 0; nt < 4; ++nt) {
      int f   = wv * 64 + nt * 16 + cl;
      s[nt]   = g2[f] * rsqrtf(v2[f] + BN_EPS);
      off[nt] = b2[f] - m2[f];
      bt[nt]  = be2[f];
      wc0[nt] = Wc[f];
      wc1[nt] = Wc[512 + f];
    }

    // wave-local partial u over this wave's 64 f-columns
#pragma unroll
    for (int mt = 0; mt < 4; ++mt) {
#pragma unroll
      for (int vv = 0; vv < 8; ++vv) {
        float p0 = 0.f, p1 = 0.f;
#pragma unroll
        for (int nt = 0; nt < 4; ++nt) {
          float y = (acc[mt][nt][vv] + off[nt]) * s[nt] + bt[nt];
          y = y > 0.f ? y : SLOPE * y;
          p0 += y * wc0[nt];
          p1 += y * wc1[nt];
        }
#pragma unroll
        for (int msk = 8; msk > 0; msk >>= 1) {
          p0 += __shfl_xor(p0, msk, 16);
          p1 += __shfl_xor(p1, msk, 16);
        }
        if (cl == 0) {
          int m = mt * 16 + vv + 8 * hf;      // 0..63, halves cover disjoint rows
          up[(wv * 64 + m) * 2 + 0] = p0;
          up[(wv * 64 + m) * 2 + 1] = p1;
        }
      }
    }
  }
  __syncthreads();

  // ---- cross-wave reduction of u partials, write u[r0..r0+63][2]
  if (tid < 128) {
    int m = tid >> 1, c = tid & 1;
    float acc = 0.f;
#pragma unroll
    for (int wvi = 0; wvi < 8; ++wvi) acc += up[(wvi * 64 + m) * 2 + c];
    u[(size_t)(r0 + m) * 2 + c] = acc;
  }
}

// ---------------- 2-channel GCN mix + bias: out[b,n,c] = bc[c] + sum_m wnf[m,n]*u[b,m,c]
__global__ void __launch_bounds__(256) k_out(
    const float* __restrict__ wnf,   // [512 m][512 n] f32
    const float* __restrict__ u,     // [128][512 m][2] f32
    const float* __restrict__ bc,    // [2]
    float* __restrict__ out)         // [65536][2] f32
{
  __shared__ float ul[1024];         // u for this batch: [512][2]
  const int b    = blockIdx.x >> 1;
  const int half = blockIdx.x & 1;
  const int tid  = threadIdx.x;

  const float* ub = u + (size_t)b * 1024;
  for (int i = tid; i < 1024; i += 256) ul[i] = ub[i];
  __syncthreads();

  const int n = half * 256 + tid;
  float a0 = bc[0], a1 = bc[1];
  for (int m = 0; m < 512; ++m) {
    float wv = wnf[m * 512 + n];     // lane-coalesced
    a0 += wv * ul[2 * m];            // broadcast from LDS
    a1 += wv * ul[2 * m + 1];
  }
  size_t row = (size_t)b * 512 + n;
  out[row * 2]     = a0;
  out[row * 2 + 1] = a1;
}

extern "C" void kernel_launch(void* const* d_in, const int* in_sizes, int n_in,
                              void* d_out, int out_size, void* d_ws, size_t ws_size,
                              hipStream_t stream) {
  const float* d   = (const float*)d_in[0];   // (128,128,1,4,512) -> 65536 x 512
  const float* w   = (const float*)d_in[1];   // 512 x 512
  const float* W1  = (const float*)d_in[2];
  const float* b1  = (const float*)d_in[3];
  const float* g1  = (const float*)d_in[4];
  const float* be1 = (const float*)d_in[5];
  const float* m1  = (const float*)d_in[6];
  const float* v1  = (const float*)d_in[7];
  const float* W2  = (const float*)d_in[8];
  const float* b2  = (const float*)d_in[9];
  const float* g2  = (const float*)d_in[10];
  const float* be2 = (const float*)d_in[11];
  const float* m2  = (const float*)d_in[12];
  const float* v2  = (const float*)d_in[13];
  const float* Wc  = (const float*)d_in[14];
  const float* bc  = (const float*)d_in[15];
  float* out = (float*)d_out;                 // 65536 x 2

  char* ws = (char*)d_ws;
  float* dinv         = (float*)ws;                            // @0        (2 KB)
  float* wnf          = (float*)(ws + 4096);                   // @4 KB     (1 MB)
  unsigned short* W1b = (unsigned short*)(ws + 4096 + 1048576);            // (512 KB)
  unsigned short* W2b = (unsigned short*)(ws + 4096 + 1048576 + 524288);   // (512 KB)
  float* u            = (float*)(ws + 4096 + 1048576 + 2 * 524288);        // (512 KB)

  k_dinv <<<512, 32, 0, stream>>>(w, dinv);
  k_wnorm<<<(512 * 512) / 256, 256, 0, stream>>>(w, dinv, wnf);
  k_cvt  <<<(512 * 512 / 4) / 256, 256, 0, stream>>>(W1, W1b, 512 * 512 / 4);
  k_cvt  <<<(512 * 512 / 4) / 256, 256, 0, stream>>>(W2, W2b, 512 * 512 / 4);
  k_mlp  <<<1024, 256, 2 * 64 * LDH * 2, stream>>>(d, W1b, b1, g1, be1, m1, v1,
                                                   W2b, b2, g2, be2, m2, v2, Wc, u);
  k_out  <<<256, 256, 0, stream>>>(wnf, u, bc, out);
}
